// pairs_of_pairs_1803886265872
// MI455X (gfx1250) — matrix operand order repaired and checked by reading, not compiled, not run on hardware
//
#include <hip/hip_runtime.h>
#include <stdint.h>

typedef __attribute__((ext_vector_type(16))) _Float16 v16h;
typedef __attribute__((ext_vector_type(8)))  _Float16 v8h;
typedef __attribute__((ext_vector_type(8)))  float    v8f;

#define NB      32      // batch
#define NS      64      // size S
#define ND      63      // S-1
#define NOC     256     // channels per layer
#define TILE    64      // pixels per workgroup (one (b,d) row of w)
#define PITCH_H 264     // act row pitch in halves (256 + 8 pad) -> 528 B, 16B multiple
#define PITCH_B 528
#define OPITCH  260     // f32 out row pitch in floats (1040 B, 16B multiple)

// ---------------------------------------------------------------------------
// Weight pre-pack: f32 W[o][c] (256x256) -> f16 WMMA B-fragments in d_ws.
// Fragment for (layer l, kstep ks, ntile nt): 32 lanes x 32B.
// Layout: dword index = l*32768 + ks*4096 + nt*256 + h*128 + lane*4 + j
//   lane: n = lane%16, half = lane/16
//   c = ks*32 + h*16 + half*8 + 2*j (+0/1 packed lo/hi), o = nt*16 + n
// Main kernel loads each fragment as two coalesced b128 (h=0 at +0, h=1 at +512B).
// ---------------------------------------------------------------------------
__global__ __launch_bounds__(256) void pack_weights_k(
    const float* __restrict__ W1, const float* __restrict__ W2,
    const float* __restrict__ W3, uint32_t* __restrict__ wpack)
{
  int tid = blockIdx.x * 256 + threadIdx.x;          // 0 .. 98303
  if (tid >= 3 * 8 * 16 * 2 * 128) return;
  int j    = tid & 3;
  int lane = (tid >> 2) & 31;
  int h    = (tid >> 7) & 1;
  int nt   = (tid >> 8) & 15;
  int ks   = (tid >> 12) & 7;
  int l    = tid >> 15;
  const float* W = (l == 0) ? W1 : (l == 1) ? W2 : W3;
  int n = lane & 15, kh = lane >> 4;
  int o = nt * 16 + n;
  int c = ks * 32 + h * 16 + kh * 8 + 2 * j;
  union { _Float16 hh[2]; uint32_t u; } pk;
  pk.hh[0] = (_Float16)W[o * 256 + c];
  pk.hh[1] = (_Float16)W[o * 256 + c + 1];
  wpack[tid] = pk.u;
}

// ---------------------------------------------------------------------------
// Fused: assemble pairs_cat tile in LDS (f16), 3 x (GEMM 64x256x256 + bias +
// ReLU) via v_wmma_f32_16x16x32_f16, intermediates ping-pong in LDS,
// final layer transposed through LDS (f32) for coalesced global stores.
// grid = 32*63 blocks, 256 threads (8 waves).
// ---------------------------------------------------------------------------
__global__ __launch_bounds__(256) void fused3_k(
    const float* __restrict__ x, const float* __restrict__ xc,
    const float* __restrict__ b1, const float* __restrict__ b2,
    const float* __restrict__ b3, const uint32_t* __restrict__ wpack,
    float* __restrict__ out)
{
  __shared__ __attribute__((aligned(16))) char smem[2 * TILE * PITCH_B]; // 67584 B

  const int tid  = threadIdx.x;
  const int lane = tid & 31;
  const int wv   = tid >> 5;
  const int g    = blockIdx.x;
  const int b    = g / ND;
  const int d    = g % ND;

  _Float16* act0 = (_Float16*)smem;

  // ---- Stage layer-0 input: act0[w][c], c = [xc(128) | x_rep(64) | rolled(64)]
  {
    const float* xcb = xc + (((size_t)b * 128) * ND + d) * NS;
    const float* xb  = x + (size_t)b * 64 * NS;
#pragma unroll
    for (int it = 0; it < 16; ++it) {          // xc: 4096 channel-pairs
      int pe = it * 256 + tid;
      int w = pe & 63, c = (pe >> 6) * 2;
      union { _Float16 hh[2]; uint32_t u; } pk;
      pk.hh[0] = (_Float16)xcb[(size_t)c * (ND * NS) + w];
      pk.hh[1] = (_Float16)xcb[(size_t)(c + 1) * (ND * NS) + w];
      *(uint32_t*)&act0[w * PITCH_H + c] = pk.u;
    }
#pragma unroll
    for (int it = 0; it < 8; ++it) {           // x broadcast -> ch 128..191
      int pe = it * 256 + tid;
      int w = pe & 63, c = (pe >> 6) * 2;
      union { _Float16 hh[2]; uint32_t u; } pk;
      pk.hh[0] = (_Float16)xb[c * NS + w];
      pk.hh[1] = (_Float16)xb[(c + 1) * NS + w];
      *(uint32_t*)&act0[w * PITCH_H + 128 + c] = pk.u;
    }
#pragma unroll
    for (int it = 0; it < 8; ++it) {           // rolled -> ch 192..255
      int pe = it * 256 + tid;
      int w = pe & 63, c = (pe >> 6) * 2;
      int ws = (w - d - 1) & 63;
      union { _Float16 hh[2]; uint32_t u; } pk;
      pk.hh[0] = (_Float16)xb[c * NS + ws];
      pk.hh[1] = (_Float16)xb[(c + 1) * NS + ws];
      *(uint32_t*)&act0[w * PITCH_H + 192 + c] = pk.u;
    }
  }
  __syncthreads();

  const int m  = lane & 15;     // M / N index within tile
  const int kh = lane >> 4;     // half

  v8f acc[4][2];

  for (int layer = 0; layer < 3; ++layer) {
    const unsigned inOff = (layer & 1) ? (unsigned)(TILE * PITCH_B) : 0u;
    const _Float16* inAct = (const _Float16*)(smem + inOff);
    const float* bias = (layer == 0) ? b1 : (layer == 1) ? b2 : b3;
    const uint32_t* wl = wpack + (size_t)layer * 32768;

    const float bv0 = bias[wv * 32 + m];
    const float bv1 = bias[wv * 32 + 16 + m];
#pragma unroll
    for (int mt = 0; mt < 4; ++mt)
#pragma unroll
      for (int i = 0; i < 8; ++i) { acc[mt][0][i] = bv0; acc[mt][1][i] = bv1; }

#pragma unroll
    for (int ks = 0; ks < 8; ++ks) {
      // A fragments: 16x32 f16, per-lane K = (j<4?0:16)+kh*8+2(j&3) -> two b128
      v16h A[4];
#pragma unroll
      for (int mt = 0; mt < 4; ++mt) {
        const _Float16* ap = inAct + (mt * 16 + m) * PITCH_H + ks * 32 + kh * 8;
        union { v16h v; v8h h2[2]; } ua;
        ua.h2[0] = *(const v8h*)ap;
        ua.h2[1] = *(const v8h*)(ap + 16);
        A[mt] = ua.v;
      }
      // B fragments: packed f16 weights, two coalesced b128 per lane
      v16h Bf[2];
#pragma unroll
      for (int nt = 0; nt < 2; ++nt) {
        const uint32_t* wp = wl + ((ks * 16 + (wv * 2 + nt)) * 2) * 128 + lane * 4;
        union { v16h v; v8h h2[2]; } ub;
        ub.h2[0] = *(const v8h*)wp;
        ub.h2[1] = *(const v8h*)(wp + 128);
        Bf[nt] = ub.v;
      }
#pragma unroll
      for (int mt = 0; mt < 4; ++mt)
#pragma unroll
        for (int nt = 0; nt < 2; ++nt)
          acc[mt][nt] = __builtin_amdgcn_wmma_f32_16x16x32_f16(
              false, A[mt], false, Bf[nt], (short)0, acc[mt][nt], false, false);
    }

    if (layer < 2) {
      // ReLU + store f16 activations for next layer
      _Float16* outAct = (_Float16*)(smem + ((layer & 1) ? 0u : (unsigned)(TILE * PITCH_B)));
#pragma unroll
      for (int mt = 0; mt < 4; ++mt)
#pragma unroll
        for (int nt = 0; nt < 2; ++nt)
#pragma unroll
          for (int i = 0; i < 8; ++i) {
            int p = mt * 16 + i + (kh ? 8 : 0);
            int c = wv * 32 + nt * 16 + m;
            outAct[p * PITCH_H + c] = (_Float16)fmaxf(acc[mt][nt][i], 0.0f);
          }
      __syncthreads();
    } else {
      // Last layer: all reads of act buffers done -> reuse smem as f32 [w][o]
      __syncthreads();
      float* ldsOut = (float*)smem;
#pragma unroll
      for (int mt = 0; mt < 4; ++mt)
#pragma unroll
        for (int nt = 0; nt < 2; ++nt)
#pragma unroll
          for (int i = 0; i < 8; ++i) {
            int p = mt * 16 + i + (kh ? 8 : 0);
            int c = wv * 32 + nt * 16 + m;
            ldsOut[p * OPITCH + c] = fmaxf(acc[mt][nt][i], 0.0f);
          }
      __syncthreads();
      // Transposed, coalesced writeout: out[((b*256+o)*63+d)*64 + w]
      float* ob = out + (((size_t)b * NOC + wv * 32) * ND + d) * NS;
      const int w0 = lane * 2;
#pragma unroll
      for (int oo = 0; oo < 32; ++oo) {
        float2 v;
        v.x = ldsOut[w0 * OPITCH + wv * 32 + oo];
        v.y = ldsOut[(w0 + 1) * OPITCH + wv * 32 + oo];
        *(float2*)(ob + (size_t)oo * (ND * NS) + w0) = v;
      }
    }
  }
}

extern "C" void kernel_launch(void* const* d_in, const int* in_sizes, int n_in,
                              void* d_out, int out_size, void* d_ws, size_t ws_size,
                              hipStream_t stream) {
  const float* x  = (const float*)d_in[0];
  const float* xc = (const float*)d_in[1];
  const float* W1 = (const float*)d_in[2];
  const float* b1 = (const float*)d_in[3];
  const float* W2 = (const float*)d_in[4];
  const float* b2 = (const float*)d_in[5];
  const float* W3 = (const float*)d_in[6];
  const float* b3 = (const float*)d_in[7];
  uint32_t* wpack = (uint32_t*)d_ws;   // 384 KB used

  pack_weights_k<<<384, 256, 0, stream>>>(W1, W2, W3, wpack);
  fused3_k<<<NB * ND, 256, 0, stream>>>(x, xc, b1, b2, b3, wpack, (float*)d_out);
}
